// MoIETransformerBlock_45990509805775
// MI455X (gfx1250) — matrix operand order repaired and checked by reading, not compiled, not run on hardware
//
#include <hip/hip_runtime.h>

#define D_MODEL 1024
#define FDIM    4096
#define SEQ     2048
#define BATCH   2

typedef __attribute__((ext_vector_type(16))) __bf16 v16bf;
typedef __attribute__((ext_vector_type(8)))  float  v8f;
typedef __attribute__((ext_vector_type(4)))  unsigned int uint4v;
typedef __attribute__((ext_vector_type(8)))  int int8v;
typedef __attribute__((ext_vector_type(4)))  int int4v;

union FragBF {
    v16bf v;
    unsigned short u[16];
    unsigned int d[8];
};

__device__ __forceinline__ unsigned short f2bf(float f) {
    unsigned int u = __float_as_uint(f);
    u += 0x7FFFu + ((u >> 16) & 1u);   // round-to-nearest-even
    return (unsigned short)(u >> 16);
}

__device__ __forceinline__ float block_reduce_sum(float v, float* red) {
    const int t = threadIdx.x;
    red[t] = v;
    __syncthreads();
    for (int o = 128; o > 0; o >>= 1) {
        if (t < o) red[t] += red[t + o];
        __syncthreads();
    }
    float r = red[0];
    __syncthreads();
    return r;
}

// ---------------------------------------------------------------------------
// Tensor Data Mover: async DMA of a 2-D bf16 tile (tile_rows x BK elems) from
// global (row stride = stride_elems) into LDS, with +1 DWORD padding after
// every 16 DWORDs => LDS row pitch of BK+2 halves. Tracked on TENSORcnt.
// ---------------------------------------------------------------------------
#define BK 32

__device__ __forceinline__ void tdm_load_2d(unsigned lds_off,
                                            const unsigned short* gptr,
                                            unsigned tensor_d0,   // row length (elems)
                                            unsigned tensor_d1,   // total rows
                                            unsigned stride_elems,
                                            unsigned tile_rows) {
    const unsigned long long ga = (unsigned long long)(size_t)gptr;
    uint4v g0;
    g0.x = 1u;                                    // count=1, user descriptor
    g0.y = lds_off;                               // LDS byte address
    g0.z = (unsigned)ga;                          // global addr [95:64]
    g0.w = (unsigned)((ga >> 32) & 0x01FFFFFFu)   // global addr [120:96]
           | 0x80000000u;                         // type=2 ("image") [127:126]
    int8v g1;
    g1[0] = (int)((1u << 16)                      // data_size = 1 -> 2 bytes
                  | (1u << 20)                    // pad_enable
                  | (3u << 22));                  // pad_interval: 16 DWORDs
                                                  // pad_amount bits = 0 -> 1 DWORD
    g1[1] = (int)((tensor_d0 & 0xFFFFu) << 16);   // tensor_dim0 lo16 (barrier addr = 0)
    g1[2] = (int)((tensor_d0 >> 16) | ((tensor_d1 & 0xFFFFu) << 16));
    g1[3] = (int)((tensor_d1 >> 16) | ((unsigned)BK << 16));  // tile_dim0 = BK
    g1[4] = (int)(tile_rows & 0xFFFFu);           // tile_dim1 (tile_dim2 = 0)
    g1[5] = (int)stride_elems;                    // tensor_dim0_stride lo32
    g1[6] = 0;
    g1[7] = 0;
    const int4v z4 = (int4v)0;
#if defined(__clang_major__) && (__clang_major__ >= 23)
    const int8v z8 = (int8v)0;
    __builtin_amdgcn_tensor_load_to_lds(g0, g1, z4, z4, z8, 0);
#else
    __builtin_amdgcn_tensor_load_to_lds(g0, g1, z4, z4, 0);
#endif
}

// ---------------------------------------------------------------------------
// elementwise fp32 -> bf16
// ---------------------------------------------------------------------------
__global__ __launch_bounds__(256) void convert_bf16_kernel(
    const float* __restrict__ in, unsigned short* __restrict__ out, long long n) {
    long long i = (long long)blockIdx.x * blockDim.x + threadIdx.x;
    long long stride = (long long)gridDim.x * blockDim.x;
    for (; i < n; i += stride) out[i] = f2bf(in[i]);
}

// ---------------------------------------------------------------------------
// per-row: bf16 copy + 1/max(||row||, 1e-12)
// ---------------------------------------------------------------------------
__global__ __launch_bounds__(256) void rownorm_convert_kernel(
    const float* __restrict__ in, unsigned short* __restrict__ outb,
    float* __restrict__ invn, int C) {
    __shared__ float red[256];
    const int r = blockIdx.x;
    const float* p = in + (size_t)r * C;
    unsigned short* q = outb + (size_t)r * C;
    float s = 0.f;
    for (int c = threadIdx.x; c < C; c += 256) {
        float v = p[c];
        s += v * v;
        q[c] = f2bf(v);
    }
    float tot = block_reduce_sum(s, red);
    if (threadIdx.x == 0) invn[r] = 1.f / fmaxf(sqrtf(tot), 1e-12f);
}

// ---------------------------------------------------------------------------
// row LayerNorm, optional base add: out = base + (x-m)/sqrt(v+eps)*w + b
// ---------------------------------------------------------------------------
__global__ __launch_bounds__(256) void layernorm_kernel(
    const float* __restrict__ in, const float* __restrict__ w,
    const float* __restrict__ b, const float* __restrict__ base,
    float* __restrict__ out, int C) {
    __shared__ float red[256];
    const int r = blockIdx.x;
    const float* p = in + (size_t)r * C;
    float s = 0.f;
    for (int c = threadIdx.x; c < C; c += 256) s += p[c];
    const float mean = block_reduce_sum(s, red) / (float)C;
    float v = 0.f;
    for (int c = threadIdx.x; c < C; c += 256) {
        float d = p[c] - mean;
        v += d * d;
    }
    const float rstd = rsqrtf(block_reduce_sum(v, red) / (float)C + 1e-5f);
    float* q = out + (size_t)r * C;
    const float* bp = base ? base + (size_t)r * C : nullptr;
    for (int c = threadIdx.x; c < C; c += 256) {
        float y = (p[c] - mean) * rstd * w[c] + b[c];
        if (bp) y += bp[c];
        q[c] = y;
    }
}

// ---------------------------------------------------------------------------
// out = a + b
// ---------------------------------------------------------------------------
__global__ __launch_bounds__(256) void add_kernel(
    const float* __restrict__ a, const float* __restrict__ b,
    float* __restrict__ out, long long n) {
    long long i = (long long)blockIdx.x * blockDim.x + threadIdx.x;
    long long stride = (long long)gridDim.x * blockDim.x;
    for (; i < n; i += stride) out[i] = a[i] + b[i];
}

// ---------------------------------------------------------------------------
// RoPE: split m_qkv -> q,k,v ; apply rotary to q,k ; emit bf16
// ---------------------------------------------------------------------------
__global__ __launch_bounds__(256) void rope_kernel(
    const float* __restrict__ mqkv, const float* __restrict__ cosb,
    const float* __restrict__ sinb, unsigned short* __restrict__ qb,
    unsigned short* __restrict__ kb, unsigned short* __restrict__ vb) {
    const int row = blockIdx.x;         // b*SEQ + s
    const int s = row & (SEQ - 1);
    const float* mp = mqkv + (size_t)row * (3 * D_MODEL);
    const float* cp = cosb + (size_t)s * D_MODEL;
    const float* sp = sinb + (size_t)s * D_MODEL;
    const int H = D_MODEL / 2;
    for (int c = threadIdx.x; c < D_MODEL; c += 256) {
        const float cv = cp[c], sv = sp[c];
        const float qv = mp[c], kv = mp[D_MODEL + c], vv = mp[2 * D_MODEL + c];
        const float qr = (c < H) ? -mp[c + H] : mp[c - H];
        const float kr = (c < H) ? -mp[D_MODEL + c + H] : mp[D_MODEL + c - H];
        const size_t o = (size_t)row * D_MODEL + c;
        qb[o] = f2bf(qv * cv + qr * sv);
        kb[o] = f2bf(kv * cv + kr * sv);
        vb[o] = f2bf(vv);
    }
}

// ---------------------------------------------------------------------------
// WMMA GEMM (C = A @ B^T), A:[M,K] bf16, B:[N,K] bf16, C:[M,N] f32.
// MODE==1: dual accumulators (B1=mu, B2=proto) + SPL epilogue
//   out = (A@B1^T + bias) * relu((A@B2^T)*rnA*rnB - gate), opt final ReLU.
// Block = 256 thr = 8 waves; tile 64(M) x 128(N); BK=32.
// Tiles streamed by the Tensor Data Mover, double buffered, TENSORcnt synced.
// ---------------------------------------------------------------------------
#define BM 64
#define BN 128
#define LP (BK + 2)

template <int MODE>
__global__ __launch_bounds__(256) void spl_gemm_kernel(
    const unsigned short* __restrict__ A, const unsigned short* __restrict__ B1,
    const unsigned short* __restrict__ B2, const float* __restrict__ rnA,
    const float* __restrict__ rnB, const float* __restrict__ bias,
    const float* __restrict__ gate, float* __restrict__ C,
    int M, int N, int K, int relu_out) {
    __shared__ unsigned short As[2][BM][LP];
    __shared__ unsigned short Bs1[2][BN][LP];
    __shared__ unsigned short Bs2[2][BN][LP];

    const int tid = threadIdx.x;
    const int lane = tid & 31;
    const int w = tid >> 5;
    const int wm = (w & 1) * 32;   // 2 waves down
    const int wn = (w >> 1) * 32;  // 4 waves across
    const int row0 = blockIdx.y * BM;
    const int col0 = blockIdx.x * BN;
    const int g8 = (lane >> 4) << 3;

    v8f acc1[2][2] = {};
    v8f acc2[2][2] = {};

    const unsigned ldsA[2]  = {(unsigned)(size_t)(void*)&As[0][0][0],
                               (unsigned)(size_t)(void*)&As[1][0][0]};
    const unsigned ldsB1[2] = {(unsigned)(size_t)(void*)&Bs1[0][0][0],
                               (unsigned)(size_t)(void*)&Bs1[1][0][0]};
    const unsigned ldsB2[2] = {(unsigned)(size_t)(void*)&Bs2[0][0][0],
                               (unsigned)(size_t)(void*)&Bs2[1][0][0]};

    auto issue = [&](int t, int buf) {
        const unsigned short* ap = A + (size_t)row0 * K + (size_t)t * BK;
        tdm_load_2d(ldsA[buf], ap, (unsigned)K, (unsigned)M, (unsigned)K, BM);
        const unsigned short* b1p = B1 + (size_t)col0 * K + (size_t)t * BK;
        tdm_load_2d(ldsB1[buf], b1p, (unsigned)K, (unsigned)N, (unsigned)K, BN);
        if (MODE) {
            const unsigned short* b2p = B2 + (size_t)col0 * K + (size_t)t * BK;
            tdm_load_2d(ldsB2[buf], b2p, (unsigned)K, (unsigned)N, (unsigned)K, BN);
        }
    };

    const int nk = K / BK;
    if (w == 0) issue(0, 0);
    int cur = 0;
    for (int t = 0; t < nk; ++t) {
        if (w == 0) __builtin_amdgcn_s_wait_tensorcnt(0);  // stage t resident
        __syncthreads();
        if (w == 0 && t + 1 < nk) issue(t + 1, cur ^ 1);   // prefetch stage t+1

        FragBF afrag[2];
#pragma unroll
        for (int tm = 0; tm < 2; ++tm) {
            const int m = wm + tm * 16 + (lane & 15);
#pragma unroll
            for (int vv = 0; vv < 8; ++vv) {
                const int kk = ((vv >> 2) << 4) + g8 + ((vv & 3) << 1);
                afrag[tm].d[vv] = *(const unsigned int*)&As[cur][m][kk];
            }
        }
#pragma unroll
        for (int tn = 0; tn < 2; ++tn) {
            FragBF b1f, b2f;
            const int n = wn + tn * 16;
#pragma unroll
            for (int h = 0; h < 16; ++h) b1f.u[h] = Bs1[cur][n + (lane & 15)][((h >> 3) << 4) + g8 + (h & 7)];
            if (MODE) {
#pragma unroll
                for (int h = 0; h < 16; ++h) b2f.u[h] = Bs2[cur][n + (lane & 15)][((h >> 3) << 4) + g8 + (h & 7)];
            }
#pragma unroll
            for (int tm = 0; tm < 2; ++tm) {
                acc1[tm][tn] = __builtin_amdgcn_wmma_f32_16x16x32_bf16(
                    false, afrag[tm].v, false, b1f.v, (short)0, acc1[tm][tn],
                    false, false);
                if (MODE)
                    acc2[tm][tn] = __builtin_amdgcn_wmma_f32_16x16x32_bf16(
                        false, afrag[tm].v, false, b2f.v, (short)0, acc2[tm][tn],
                        false, false);
            }
        }
        cur ^= 1;
    }

    const int moff = (lane >> 4) * 8;
    const int nlane = lane & 15;
#pragma unroll
    for (int tm = 0; tm < 2; ++tm)
#pragma unroll
        for (int tn = 0; tn < 2; ++tn) {
#pragma unroll
            for (int j = 0; j < 8; ++j) {
                const int r = row0 + wm + tm * 16 + moff + j;
                const int c = col0 + wn + tn * 16 + nlane;
                float out;
                if (MODE) {
                    const float comp = acc1[tm][tn][j] + bias[c];
                    const float sc = acc2[tm][tn][j] * rnA[r] * rnB[c];
                    const float g = sc - gate[c];
                    out = comp * (g > 0.f ? g : 0.f);
                } else {
                    out = acc1[tm][tn][j];
                }
                if (relu_out) out = fmaxf(out, 0.f);
                C[(size_t)r * N + c] = out;
            }
        }
}

// ---------------------------------------------------------------------------
// Causal attention, 16 query rows / workgroup, score strip resident in LDS.
// Phase 1: S = (Q K^T)/32 via WMMA.  Phase 2: softmax (wave shuffles).
// Phase 3: O = P V via WMMA, V tiles staged cooperatively.
// ---------------------------------------------------------------------------
#define SPITCH (SEQ + 1)
#define QPITCH (D_MODEL + 2)

__global__ __launch_bounds__(256) void attention_kernel(
    const unsigned short* __restrict__ qb, const unsigned short* __restrict__ kb,
    const unsigned short* __restrict__ vb, float* __restrict__ attn_out) {
    extern __shared__ char smem[];
    float* sc = (float*)smem;                                     // [16][SPITCH]
    unsigned short* Qs =
        (unsigned short*)(smem + 16 * SPITCH * sizeof(float));    // [16][QPITCH]
    unsigned short* Vt = Qs + 16 * QPITCH;                        // [32][QPITCH]

    const int qt = blockIdx.x;   // query tile (16 rows)
    const int bi = blockIdx.y;   // batch
    const int tid = threadIdx.x;
    const int lane = tid & 31;
    const int w = tid >> 5;
    const size_t rowbase = (size_t)bi * SEQ + (size_t)qt * 16;
    const int g8 = (lane >> 4) << 3;

    // stage Q block (16 x D)
    for (int i = tid; i < 16 * D_MODEL; i += 256) {
        const int r = i >> 10, c = i & (D_MODEL - 1);
        Qs[r * QPITCH + c] = qb[(rowbase + r) * D_MODEL + c];
    }
    __syncthreads();

    // ---- phase 1: scores ----
    for (int kt = w; kt <= qt; kt += 8) {
        v8f acc = {};
        const unsigned short* kbase =
            kb + ((size_t)bi * SEQ + (size_t)kt * 16) * D_MODEL;
        for (int d0 = 0; d0 < D_MODEL; d0 += 32) {
            FragBF a, bfr;
            const int m = lane & 15;
#pragma unroll
            for (int vv = 0; vv < 8; ++vv) {
                const int kk = ((vv >> 2) << 4) + g8 + ((vv & 3) << 1);
                a.d[vv] = *(const unsigned int*)&Qs[m * QPITCH + d0 + kk];
            }
            const unsigned short* kp = kbase + (size_t)(lane & 15) * D_MODEL + d0;
#pragma unroll
            for (int h = 0; h < 16; ++h) bfr.u[h] = kp[((h >> 3) << 4) + g8 + (h & 7)];
            acc = __builtin_amdgcn_wmma_f32_16x16x32_bf16(
                false, a.v, false, bfr.v, (short)0, acc, false, false);
        }
        const int key = kt * 16 + (lane & 15);
#pragma unroll
        for (int j = 0; j < 8; ++j) {
            const int m = ((lane >> 4) << 3) + j;
            const int qg = qt * 16 + m;
            sc[m * SPITCH + key] = (key <= qg) ? acc[j] * 0.03125f : -1e30f;
        }
    }
    __syncthreads();

    // ---- phase 2: softmax, 2 rows per wave ----
    const int kmax = (qt + 1) * 16;
    const int kpad = (kmax + 31) & ~31;
    for (int rr = 0; rr < 2; ++rr) {
        const int r = w * 2 + rr;
        const int qg = qt * 16 + r;
        float mx = -1e30f;
        for (int c = lane; c <= qg; c += 32) mx = fmaxf(mx, sc[r * SPITCH + c]);
        for (int o = 16; o > 0; o >>= 1) mx = fmaxf(mx, __shfl_xor(mx, o, 32));
        float sum = 0.f;
        for (int c = lane; c <= qg; c += 32) {
            const float e = __expf(sc[r * SPITCH + c] - mx);
            sc[r * SPITCH + c] = e;
            sum += e;
        }
        for (int o = 16; o > 0; o >>= 1) sum += __shfl_xor(sum, o, 32);
        const float inv = 1.f / sum;
        for (int c = lane; c < kpad; c += 32)
            sc[r * SPITCH + c] = (c <= qg) ? sc[r * SPITCH + c] * inv : 0.f;
    }
    __syncthreads();

    // ---- phase 3: O = P @ V (each wave owns 128 output cols) ----
    v8f acc[8] = {};
    for (int kk0 = 0; kk0 < kmax; kk0 += 32) {
        __syncthreads();
        for (int i = tid; i < 32 * D_MODEL; i += 256) {
            const int r = i >> 10, c = i & (D_MODEL - 1);
            const int key = kk0 + r;
            Vt[r * QPITCH + c] =
                (key < kmax)
                    ? vb[((size_t)bi * SEQ + key) * D_MODEL + c]
                    : (unsigned short)0;
        }
        __syncthreads();
        FragBF a;
        {
            const int m = lane & 15;
#pragma unroll
            for (int vv = 0; vv < 8; ++vv) {
                const int kk = ((vv >> 2) << 4) + g8 + ((vv & 3) << 1);
                a.u[2 * vv] = f2bf(sc[m * SPITCH + kk0 + kk]);
                a.u[2 * vv + 1] = f2bf(sc[m * SPITCH + kk0 + kk + 1]);
            }
        }
#pragma unroll
        for (int t = 0; t < 8; ++t) {
            FragBF bfr;
            const int n0 = w * 128 + t * 16;
#pragma unroll
            for (int h = 0; h < 16; ++h) bfr.u[h] = Vt[(((h >> 3) << 4) + g8 + (h & 7)) * QPITCH + n0 + (lane & 15)];
            acc[t] = __builtin_amdgcn_wmma_f32_16x16x32_bf16(
                false, a.v, false, bfr.v, (short)0, acc[t], false, false);
        }
    }

#pragma unroll
    for (int t = 0; t < 8; ++t) {
        const int n = w * 128 + t * 16 + (lane & 15);
#pragma unroll
        for (int j = 0; j < 8; ++j) {
            const int m = ((lane >> 4) << 3) + j;
            attn_out[(rowbase + m) * D_MODEL + n] = acc[t][j];
        }
    }
}

// ---------------------------------------------------------------------------
// host orchestration
// ---------------------------------------------------------------------------
extern "C" void kernel_launch(void* const* d_in, const int* in_sizes, int n_in,
                              void* d_out, int out_size, void* d_ws,
                              size_t ws_size, hipStream_t stream) {
    (void)in_sizes; (void)n_in; (void)out_size; (void)ws_size;
    const float* x        = (const float*)d_in[0];
    const float* cosb     = (const float*)d_in[1];
    const float* sinb     = (const float*)d_in[2];
    const float* ln1_w    = (const float*)d_in[3];
    const float* ln1_b    = (const float*)d_in[4];
    const float* ln2_w    = (const float*)d_in[5];
    const float* ln2_b    = (const float*)d_in[6];
    const float* qkv_mu   = (const float*)d_in[7];
    const float* qkv_pr   = (const float*)d_in[8];
    const float* qkv_bias = (const float*)d_in[9];
    const float* qkv_gate = (const float*)d_in[10];
    const float* o_mu     = (const float*)d_in[11];
    const float* o_pr     = (const float*)d_in[12];
    const float* o_bias   = (const float*)d_in[13];
    const float* o_gate   = (const float*)d_in[14];
    const float* f1_mu    = (const float*)d_in[15];
    const float* f1_pr    = (const float*)d_in[16];
    const float* f1_bias  = (const float*)d_in[17];
    const float* f1_gate  = (const float*)d_in[18];
    const float* f2_mu    = (const float*)d_in[19];
    const float* f2_pr    = (const float*)d_in[20];
    const float* f2_bias  = (const float*)d_in[21];
    const float* f2_gate  = (const float*)d_in[22];
    const float* pt_qkv   = (const float*)d_in[23];
    const float* pt_o     = (const float*)d_in[24];
    const float* pt_f1    = (const float*)d_in[25];
    const float* pt_f2    = (const float*)d_in[26];
    const float* plnq_w   = (const float*)d_in[27];
    const float* plnq_b   = (const float*)d_in[28];
    const float* plno_w   = (const float*)d_in[29];
    const float* plno_b   = (const float*)d_in[30];
    const float* plnf1_w  = (const float*)d_in[31];
    const float* plnf1_b  = (const float*)d_in[32];
    const float* plnf2_w  = (const float*)d_in[33];
    const float* plnf2_b  = (const float*)d_in[34];
    const float* prev_qkv = (const float*)d_in[35];
    const float* prev_o   = (const float*)d_in[36];
    const float* prev_f1  = (const float*)d_in[37];
    const float* prev_f2  = (const float*)d_in[38];

    const int M = BATCH * SEQ;  // 4096
    char* ws = (char*)d_ws;
    size_t off = 0;
    auto alloc = [&](size_t bytes) -> void* {
        void* p = ws + off;
        off = (off + bytes + 255) & ~(size_t)255;
        return p;
    };
    typedef unsigned short us;

    // bf16 weight copies
    us* prevq_b = (us*)alloc((size_t)3 * D_MODEL * D_MODEL * 2);
    us* ptq_b   = (us*)alloc((size_t)D_MODEL * D_MODEL * 2);
    us* prevo_b = (us*)alloc((size_t)D_MODEL * D_MODEL * 2);
    us* pto_b   = (us*)alloc((size_t)D_MODEL * D_MODEL * 2);
    us* prevf1_b= (us*)alloc((size_t)FDIM * D_MODEL * 2);
    us* ptf1_b  = (us*)alloc((size_t)D_MODEL * D_MODEL * 2);
    us* prevf2_b= (us*)alloc((size_t)D_MODEL * FDIM * 2);
    us* ptf2_b  = (us*)alloc((size_t)FDIM * FDIM * 2);
    us* muq_b   = (us*)alloc((size_t)3 * D_MODEL * D_MODEL * 2);
    us* muo_b   = (us*)alloc((size_t)D_MODEL * D_MODEL * 2);
    us* muf1_b  = (us*)alloc((size_t)FDIM * D_MODEL * 2);
    us* muf2_b  = (us*)alloc((size_t)D_MODEL * FDIM * 2);
    // proto transform tmp + eff
    float* tmpq  = (float*)alloc((size_t)3 * D_MODEL * D_MODEL * 4);
    float* tmpo  = (float*)alloc((size_t)D_MODEL * D_MODEL * 4);
    float* tmpf1 = (float*)alloc((size_t)FDIM * D_MODEL * 4);
    float* tmpf2 = (float*)alloc((size_t)D_MODEL * FDIM * 4);
    float* effq  = (float*)alloc((size_t)3 * D_MODEL * D_MODEL * 4);
    float* effo  = (float*)alloc((size_t)D_MODEL * D_MODEL * 4);
    float* efff1 = (float*)alloc((size_t)FDIM * D_MODEL * 4);
    float* efff2 = (float*)alloc((size_t)D_MODEL * FDIM * 4);
    us* effq_b   = (us*)alloc((size_t)3 * D_MODEL * D_MODEL * 2);
    us* effo_b   = (us*)alloc((size_t)D_MODEL * D_MODEL * 2);
    us* efff1_b  = (us*)alloc((size_t)FDIM * D_MODEL * 2);
    us* efff2_b  = (us*)alloc((size_t)D_MODEL * FDIM * 2);
    float* rn_effq  = (float*)alloc((size_t)3 * D_MODEL * 4);
    float* rn_effo  = (float*)alloc((size_t)D_MODEL * 4);
    float* rn_efff1 = (float*)alloc((size_t)FDIM * 4);
    float* rn_efff2 = (float*)alloc((size_t)D_MODEL * 4);
    // activations
    float* attn_in  = (float*)alloc((size_t)M * D_MODEL * 4);
    us* attnin_b    = (us*)alloc((size_t)M * D_MODEL * 2);
    float* rn_ain   = (float*)alloc((size_t)M * 4);
    float* m_qkv    = (float*)alloc((size_t)M * 3 * D_MODEL * 4);
    us* q_b = (us*)alloc((size_t)M * D_MODEL * 2);
    us* k_b = (us*)alloc((size_t)M * D_MODEL * 2);
    us* v_b = (us*)alloc((size_t)M * D_MODEL * 2);
    float* attn_out = (float*)alloc((size_t)M * D_MODEL * 4);
    us* attnout_b   = (us*)alloc((size_t)M * D_MODEL * 2);
    float* rn_aout  = (float*)alloc((size_t)M * 4);
    float* m_o      = (float*)alloc((size_t)M * D_MODEL * 4);
    float* x1       = (float*)alloc((size_t)M * D_MODEL * 4);
    float* ffn_in   = (float*)alloc((size_t)M * D_MODEL * 4);
    us* ffnin_b     = (us*)alloc((size_t)M * D_MODEL * 2);
    float* rn_fin   = (float*)alloc((size_t)M * 4);
    float* hbuf     = (float*)alloc((size_t)M * FDIM * 4);
    us* h_b         = (us*)alloc((size_t)M * FDIM * 2);
    float* rn_h     = (float*)alloc((size_t)M * 4);
    float* m2       = (float*)alloc((size_t)M * D_MODEL * 4);

    auto conv = [&](const float* in, us* out, long long n) {
        convert_bf16_kernel<<<4096, 256, 0, stream>>>(in, out, n);
    };
    auto gemm0 = [&](const us* A, const us* B1, float* C, int Mm, int Nn, int Kk) {
        spl_gemm_kernel<0><<<dim3(Nn / BN, Mm / BM), 256, 0, stream>>>(
            A, B1, nullptr, nullptr, nullptr, nullptr, nullptr, C, Mm, Nn, Kk, 0);
    };
    auto gemm1 = [&](const us* A, const us* B1, const us* B2, const float* rA,
                     const float* rB, const float* bi, const float* ga, float* C,
                     int Mm, int Nn, int Kk, int relu) {
        spl_gemm_kernel<1><<<dim3(Nn / BN, Mm / BM), 256, 0, stream>>>(
            A, B1, B2, rA, rB, bi, ga, C, Mm, Nn, Kk, relu);
    };

    // ---- weight conversions ----
    conv(prev_qkv, prevq_b, (long long)3 * D_MODEL * D_MODEL);
    conv(pt_qkv, ptq_b, (long long)D_MODEL * D_MODEL);
    conv(prev_o, prevo_b, (long long)D_MODEL * D_MODEL);
    conv(pt_o, pto_b, (long long)D_MODEL * D_MODEL);
    conv(prev_f1, prevf1_b, (long long)FDIM * D_MODEL);
    conv(pt_f1, ptf1_b, (long long)D_MODEL * D_MODEL);
    conv(prev_f2, prevf2_b, (long long)D_MODEL * FDIM);
    conv(pt_f2, ptf2_b, (long long)FDIM * FDIM);
    conv(qkv_mu, muq_b, (long long)3 * D_MODEL * D_MODEL);
    conv(o_mu, muo_b, (long long)D_MODEL * D_MODEL);
    conv(f1_mu, muf1_b, (long long)FDIM * D_MODEL);
    conv(f2_mu, muf2_b, (long long)D_MODEL * FDIM);

    // ---- effective protos: eff = proto + LN(prev @ pt^T) ----
    gemm0(prevq_b, ptq_b, tmpq, 3 * D_MODEL, D_MODEL, D_MODEL);
    gemm0(prevo_b, pto_b, tmpo, D_MODEL, D_MODEL, D_MODEL);
    gemm0(prevf1_b, ptf1_b, tmpf1, FDIM, D_MODEL, D_MODEL);
    gemm0(prevf2_b, ptf2_b, tmpf2, D_MODEL, FDIM, FDIM);
    layernorm_kernel<<<3 * D_MODEL, 256, 0, stream>>>(tmpq, plnq_w, plnq_b, qkv_pr, effq, D_MODEL);
    layernorm_kernel<<<D_MODEL, 256, 0, stream>>>(tmpo, plno_w, plno_b, o_pr, effo, D_MODEL);
    layernorm_kernel<<<FDIM, 256, 0, stream>>>(tmpf1, plnf1_w, plnf1_b, f1_pr, efff1, D_MODEL);
    layernorm_kernel<<<D_MODEL, 256, 0, stream>>>(tmpf2, plnf2_w, plnf2_b, f2_pr, efff2, FDIM);
    rownorm_convert_kernel<<<3 * D_MODEL, 256, 0, stream>>>(effq, effq_b, rn_effq, D_MODEL);
    rownorm_convert_kernel<<<D_MODEL, 256, 0, stream>>>(effo, effo_b, rn_effo, D_MODEL);
    rownorm_convert_kernel<<<FDIM, 256, 0, stream>>>(efff1, efff1_b, rn_efff1, D_MODEL);
    rownorm_convert_kernel<<<D_MODEL, 256, 0, stream>>>(efff2, efff2_b, rn_efff2, FDIM);

    // ---- attention path ----
    layernorm_kernel<<<M, 256, 0, stream>>>(x, ln1_w, ln1_b, nullptr, attn_in, D_MODEL);
    rownorm_convert_kernel<<<M, 256, 0, stream>>>(attn_in, attnin_b, rn_ain, D_MODEL);
    gemm1(attnin_b, muq_b, effq_b, rn_ain, rn_effq, qkv_bias, qkv_gate, m_qkv,
          M, 3 * D_MODEL, D_MODEL, 0);
    rope_kernel<<<M, 256, 0, stream>>>(m_qkv, cosb, sinb, q_b, k_b, v_b);

    const size_t att_lds = (size_t)16 * SPITCH * sizeof(float) +
                           (size_t)16 * QPITCH * 2 + (size_t)32 * QPITCH * 2;
    hipFuncSetAttribute((const void*)attention_kernel,
                        hipFuncAttributeMaxDynamicSharedMemorySize, (int)att_lds);
    attention_kernel<<<dim3(SEQ / 16, BATCH), 256, att_lds, stream>>>(
        q_b, k_b, v_b, attn_out);

    rownorm_convert_kernel<<<M, 256, 0, stream>>>(attn_out, attnout_b, rn_aout, D_MODEL);
    gemm1(attnout_b, muo_b, effo_b, rn_aout, rn_effo, o_bias, o_gate, m_o,
          M, D_MODEL, D_MODEL, 0);
    add_kernel<<<4096, 256, 0, stream>>>(x, m_o, x1, (long long)M * D_MODEL);

    // ---- ffn path ----
    layernorm_kernel<<<M, 256, 0, stream>>>(x1, ln2_w, ln2_b, nullptr, ffn_in, D_MODEL);
    rownorm_convert_kernel<<<M, 256, 0, stream>>>(ffn_in, ffnin_b, rn_fin, D_MODEL);
    gemm1(ffnin_b, muf1_b, efff1_b, rn_fin, rn_efff1, f1_bias, f1_gate, hbuf,
          M, FDIM, D_MODEL, /*relu*/ 1);
    rownorm_convert_kernel<<<M, 256, 0, stream>>>(hbuf, h_b, rn_h, FDIM);
    gemm1(h_b, muf2_b, efff2_b, rn_h, rn_efff2, f2_bias, f2_gate, m2,
          M, D_MODEL, FDIM, 0);
    add_kernel<<<4096, 256, 0, stream>>>(x1, m2, (float*)d_out, (long long)M * D_MODEL);
}
